// SSMBlock_45535243272948
// MI455X (gfx1250) — matrix operand order repaired and checked by reading, not compiled, not run on hardware
//
#include <hip/hip_runtime.h>

// ---------------------------------------------------------------------------
// SSM block for MI455X (gfx1250, wave32).
// Big GEMMs: v_wmma_f32_16x16x32_bf16 (bf16 in, f32 accumulate), fed by
// GLOBAL_LOAD_ASYNC_TO_LDS_B128 (ASYNCcnt-tracked DMA) in a 3-buffer,
// depth-2 pipeline. Scan / norm / conv stay fp32 (tiny, bandwidth-bound).
// ---------------------------------------------------------------------------

#define HDIM   2560
#define DINNER 8192
#define GN     32
#define DGATE  4096
#define KCONV  4
#define DHD    128
#define BSZ    2
#define LSEQ   4096
#define MROWS  (BSZ * LSEQ)   // 8192

typedef __attribute__((ext_vector_type(16))) __bf16 v16bf;
typedef __attribute__((ext_vector_type(8)))  float  v8f;
typedef int v4i __attribute__((vector_size(16)));   // matches builtin param type

struct U128x2 { uint4 lo, hi; };   // 32 bytes, bit_cast source for v16bf

// ---- CDNA5 async global->LDS copy (probe via __has_builtin) ---------------
#if defined(__AMDGCN__) && __has_builtin(__builtin_amdgcn_global_load_async_to_lds_b128)
#define HAVE_ASYNC_LDS 1
#else
#define HAVE_ASYNC_LDS 0
#endif

#if HAVE_ASYNC_LDS
#if __has_builtin(__builtin_amdgcn_s_wait_asynccnt)
#define WAIT_ASYNC(n) __builtin_amdgcn_s_wait_asynccnt(n)
#else
#define WAIT_ASYNC(n) asm volatile("s_wait_asynccnt %0" ::"i"(n) : "memory")
#endif
#else
#define WAIT_ASYNC(n)
#endif

static __device__ __forceinline__ void copy16_g2l(const unsigned short* g,
                                                  unsigned short* l) {
#if HAVE_ASYNC_LDS
  __builtin_amdgcn_global_load_async_to_lds_b128(
      (__attribute__((address_space(1))) v4i*)(g),
      (__attribute__((address_space(3))) v4i*)(l), 0, 0);
#else
  *reinterpret_cast<uint4*>(l) = *reinterpret_cast<const uint4*>(g);
#endif
}

static __device__ __forceinline__ unsigned short f2bf(float f) {
  unsigned int u = __float_as_uint(f);
  u += 0x7fffu + ((u >> 16) & 1u);       // round to nearest even
  return (unsigned short)(u >> 16);
}

// ---------------------------------------------------------------------------
// fp32 -> bf16 conversion
// ---------------------------------------------------------------------------
__global__ __launch_bounds__(256) void cvt_f32_bf16(const float* __restrict__ in,
                                                    unsigned short* __restrict__ out,
                                                    size_t n) {
  size_t i = (size_t)blockIdx.x * 256 + threadIdx.x;
  if (i < n) out[i] = f2bf(in[i]);
}

// ---------------------------------------------------------------------------
// B_coeff / C_coeff: [8192 rows] x [32+32 outputs], K = 2560, fp32.
// ---------------------------------------------------------------------------
__global__ __launch_bounds__(64) void coeff_kernel(const float* __restrict__ hs,
                                                   const float* __restrict__ Wb,
                                                   const float* __restrict__ Wa,
                                                   float* __restrict__ Bc,
                                                   float* __restrict__ Cc) {
  const int row = blockIdx.x;                  // b*L + l
  const int t   = threadIdx.x;
  const float* h = hs + (size_t)row * HDIM;
  const float* W = (t < 32) ? (Wb + (size_t)t * HDIM)
                            : (Wa + (size_t)(t - 32) * HDIM);
  float s = 0.f;
  const float4* h4 = reinterpret_cast<const float4*>(h);
  const float4* w4 = reinterpret_cast<const float4*>(W);
  #pragma unroll 4
  for (int k = 0; k < HDIM / 4; ++k) {
    float4 a = h4[k], b = w4[k];
    s += a.x * b.x + a.y * b.y + a.z * b.z + a.w * b.w;
  }
  if (t < 32) Bc[(size_t)row * GN + t] = s;
  else        Cc[(size_t)row * GN + (t - 32)] = s;
}

// ---------------------------------------------------------------------------
// WMMA bf16 GEMM:  C[M x N] (f32) = A[M x K] (bf16, row-major)
//                               x  B'[N x K] (bf16, row-major; used as B[k][n])
// Block tile 128x128x32, 256 threads (8 waves, 4x2 wave grid), each wave
// computes 2x4 WMMA 16x16 tiles. Global->LDS via async DMA, 3 LDS buffers,
// depth-2 pipeline (copies for it+2 in flight while computing it).
// Each K-step = 4 async-load instructions per wave -> s_wait_asynccnt<=4
// guarantees the oldest group has landed (loads complete in order).
// M, N divisible by 128; K divisible by 32; K/32 >= 2.
// ---------------------------------------------------------------------------
#define BM 128
#define BN 128
#define BK 32
#define LDSS 40   // padded bf16 row stride (80B) -> conflict-spread ds_load_b128

__global__ __launch_bounds__(256) void gemm_bf16_wmma(
    const unsigned short* __restrict__ A,
    const unsigned short* __restrict__ Bw,
    float* __restrict__ C,
    int M, int N, int K) {
  __shared__ __align__(16) unsigned short sA[3][BM * LDSS];
  __shared__ __align__(16) unsigned short sB[3][BN * LDSS];

  const int tid   = threadIdx.x;
  const int lane  = tid & 31;
  const int wave  = tid >> 5;
  const int lrow  = lane & 15;     // N index within tile / M row select
  const int lhalf = lane >> 4;     // 0/1
  const int wm    = (wave & 3) * 32;   // wave M offset (4 waves)
  const int wn    = (wave >> 2) * 64;  // wave N offset (2 waves)

  const int bm = blockIdx.y * BM;
  const int bn = blockIdx.x * BN;

  // staging: each thread moves 16 bf16 (two b128 DMAs) for A and for B
  const int ldrow = tid >> 1;           // 0..127
  const int ldcol = (tid & 1) * 16;     // 0 or 16
  const unsigned short* gA = A + (size_t)(bm + ldrow) * K + ldcol;
  const unsigned short* gB = Bw + (size_t)(bn + ldrow) * K + ldcol;
  const int lidx = ldrow * LDSS + ldcol;

  v8f acc[2][4];
  #pragma unroll
  for (int i = 0; i < 2; ++i)
    #pragma unroll
    for (int j = 0; j < 4; ++j)
      #pragma unroll
      for (int e = 0; e < 8; ++e) acc[i][j][e] = 0.f;

  const int niter = K / BK;

  // prologue: groups 0 and 1 in flight
  {
    copy16_g2l(gA, &sA[0][lidx]);
    copy16_g2l(gA + 8, &sA[0][lidx + 8]);
    copy16_g2l(gB, &sB[0][lidx]);
    copy16_g2l(gB + 8, &sB[0][lidx + 8]);
    copy16_g2l(gA + BK, &sA[1][lidx]);
    copy16_g2l(gA + BK + 8, &sA[1][lidx + 8]);
    copy16_g2l(gB + BK, &sB[1][lidx]);
    copy16_g2l(gB + BK + 8, &sB[1][lidx + 8]);
  }

  int cur = 0;              // buffer holding iteration 'it'
  for (int it = 0; it < niter; ++it) {
    // wait for the oldest in-flight group (this iteration's data)
    if (it + 1 < niter) { WAIT_ASYNC(4); } else { WAIT_ASYNC(0); }
    __syncthreads();

    // issue copies for it+2 into the buffer freed at it-1
    if (it + 2 < niter) {
      const int nb = (cur + 2 >= 3) ? cur - 1 : cur + 2;
      const int k0 = (it + 2) * BK;
      copy16_g2l(gA + k0, &sA[nb][lidx]);
      copy16_g2l(gA + k0 + 8, &sA[nb][lidx + 8]);
      copy16_g2l(gB + k0, &sB[nb][lidx]);
      copy16_g2l(gB + k0 + 8, &sB[nb][lidx + 8]);
    }

    // fragments: per-lane contiguous K-runs from LDS (ds_load_b128 pairs)
    v16bf afrag[2], bfrag[4];
    #pragma unroll
    for (int i = 0; i < 2; ++i) {
      const int r = wm + i * 16 + lrow;
      const uint4* p = reinterpret_cast<const uint4*>(&sA[cur][r * LDSS]);
      U128x2 t;
      t.lo = p[lhalf];       // K = lhalf*8 .. +7
      t.hi = p[2 + lhalf];   // K = 16 + lhalf*8 .. +7
      afrag[i] = __builtin_bit_cast(v16bf, t);
    }
    #pragma unroll
    for (int j = 0; j < 4; ++j) {
      const int r = wn + j * 16 + lrow;
      const uint4* p =
          reinterpret_cast<const uint4*>(&sB[cur][r * LDSS]);
      U128x2 t;
      t.lo = p[lhalf];       // K = lhalf*8 .. +7
      t.hi = p[2 + lhalf];   // K = 16 + lhalf*8 .. +7
      bfrag[j] = __builtin_bit_cast(v16bf, t);
    }

    #pragma unroll
    for (int i = 0; i < 2; ++i)
      #pragma unroll
      for (int j = 0; j < 4; ++j)
        acc[i][j] = __builtin_amdgcn_wmma_f32_16x16x32_bf16(
            false, afrag[i], false, bfrag[j], (short)0, acc[i][j],
            false, false);

    cur = (cur + 1 >= 3) ? 0 : cur + 1;
  }

  // C/D layout: VGPR e -> M = e + 8*lhalf ; N = lrow
  #pragma unroll
  for (int i = 0; i < 2; ++i) {
    const int m0 = bm + wm + i * 16 + lhalf * 8;
    #pragma unroll
    for (int j = 0; j < 4; ++j) {
      const int n0 = bn + wn + j * 16 + lrow;
      #pragma unroll
      for (int e = 0; e < 8; ++e)
        C[(size_t)(m0 + e) * N + n0] = acc[i][j][e];
    }
  }
}

// ---------------------------------------------------------------------------
// Fused causal depthwise conv (x half) + SiLU + SSM scan (scalar state).
// One thread per (b, channel c in [0,4096)); sequential over L.
// ---------------------------------------------------------------------------
__global__ __launch_bounds__(256) void scan_kernel(
    const float* __restrict__ xz, const float* __restrict__ conv_w,
    const float* __restrict__ Bc, const float* __restrict__ Cc,
    const float* __restrict__ A_log, const float* __restrict__ dt_bias,
    float* __restrict__ Y) {
  const int c = blockIdx.x * 256 + threadIdx.x;   // 0..4095
  const int b = blockIdx.y;
  const int g = c >> 7;

  const float dt = log1pf(expf(dt_bias[g]));        // softplus
  const float dA = expf(-expf(A_log[g]) * dt);      // exp(dt * (-exp(A_log)))

  const float* w = conv_w + (size_t)c * KCONV;
  const float w0 = w[0], w1 = w[1], w2 = w[2], w3 = w[3];

  const float* xzr = xz + (size_t)b * LSEQ * DINNER + c;
  const float* bcr = Bc + (size_t)b * LSEQ * GN + g;
  const float* ccr = Cc + (size_t)b * LSEQ * GN + g;
  float* yr = Y + (size_t)b * LSEQ * DGATE + c;

  float x1 = 0.f, x2 = 0.f, x3 = 0.f, h = 0.f;
  for (int l = 0; l < LSEQ; ++l) {
    const float cur = xzr[(size_t)l * DINNER];
    const float s = w3 * cur + w2 * x1 + w1 * x2 + w0 * x3;
    x3 = x2; x2 = x1; x1 = cur;
    const float xs = s / (1.f + expf(-s));          // silu
    const float bt = bcr[(size_t)l * GN];
    const float ct = ccr[(size_t)l * GN];
    h = dA * h + (dt * bt) * xs;
    yr[(size_t)l * DGATE] = ct * h;
  }
}

// ---------------------------------------------------------------------------
// RMSNorm over DH=128 + z-branch conv + SiLU gate + bf16 pack.
// Block 256 = 2 groups of 128 threads; grid = B*L*G/2.
// ---------------------------------------------------------------------------
__global__ __launch_bounds__(256) void norm_gate_kernel(
    const float* __restrict__ Y, const float* __restrict__ xz,
    const float* __restrict__ conv_w, const float* __restrict__ norm_w,
    unsigned short* __restrict__ yg) {
  __shared__ float red[8];
  const int half = threadIdx.x >> 7;                // group within block
  const int dh   = threadIdx.x & 127;
  const size_t gi = (size_t)blockIdx.x * 2 + half;  // (b*L + l)*G + g
  const int g = (int)(gi & (GN - 1));
  const size_t rowbl = gi >> 5;                     // b*L + l
  const int l = (int)(rowbl & (LSEQ - 1));

  const size_t yidx = rowbl * DGATE + (size_t)g * DHD + dh;
  const float y = Y[yidx];

  // sum of squares over 128: wave32 shuffle then 4-wave LDS combine per half
  float ss = y * y;
  #pragma unroll
  for (int off = 16; off > 0; off >>= 1) ss += __shfl_xor(ss, off);
  if ((threadIdx.x & 31) == 0) red[threadIdx.x >> 5] = ss;
  __syncthreads();
  const float tot = red[half * 4 + 0] + red[half * 4 + 1] +
                    red[half * 4 + 2] + red[half * 4 + 3];
  const float yn = y * rsqrtf(tot * (1.f / DHD) + 1e-6f) * norm_w[dh];

  // z branch: causal 4-tap conv on channel 4096 + g*128 + dh, then SiLU
  const int cz = DGATE + g * DHD + dh;
  const size_t base = rowbl * DINNER + cz;
  const float* wv = conv_w + (size_t)cz * KCONV;
  float s = wv[3] * xz[base];
  if (l >= 1) s += wv[2] * xz[base - DINNER];
  if (l >= 2) s += wv[1] * xz[base - 2 * (size_t)DINNER];
  if (l >= 3) s += wv[0] * xz[base - 3 * (size_t)DINNER];
  const float zg = s / (1.f + expf(-s));

  yg[rowbl * DGATE + (size_t)g * DHD + dh] = f2bf(yn * zg);
}

// ---------------------------------------------------------------------------
// launch
// ---------------------------------------------------------------------------
extern "C" void kernel_launch(void* const* d_in, const int* in_sizes, int n_in,
                              void* d_out, int out_size, void* d_ws,
                              size_t ws_size, hipStream_t stream) {
  const float* hs    = (const float*)d_in[0];
  const float* Wqkv  = (const float*)d_in[1];
  const float* Wb    = (const float*)d_in[2];
  const float* Wa    = (const float*)d_in[3];
  const float* convw = (const float*)d_in[4];
  const float* Wout  = (const float*)d_in[5];
  const float* normw = (const float*)d_in[6];
  const float* Alog  = (const float*)d_in[7];
  const float* dtb   = (const float*)d_in[8];
  float* out = (float*)d_out;

  // workspace layout (bytes, 256-aligned)
  char* ws = (char*)d_ws;
  size_t off = 0;
  auto take = [&](size_t bytes) {
    void* p = ws + off;
    off += (bytes + 255) & ~(size_t)255;
    return p;
  };
  unsigned short* hsb   = (unsigned short*)take((size_t)MROWS * HDIM * 2);
  unsigned short* wqkvb = (unsigned short*)take((size_t)DINNER * HDIM * 2);
  unsigned short* woutb = (unsigned short*)take((size_t)HDIM * DGATE * 2);
  float* xz = (float*)take((size_t)MROWS * DINNER * 4);
  float* Bc = (float*)take((size_t)MROWS * GN * 4);
  float* Cc = (float*)take((size_t)MROWS * GN * 4);
  float* Yb = (float*)take((size_t)MROWS * DGATE * 4);
  unsigned short* yg = (unsigned short*)take((size_t)MROWS * DGATE * 2);
  (void)ws_size; (void)in_sizes; (void)n_in; (void)out_size;

  // 1) bf16 conversions
  {
    size_t n1 = (size_t)MROWS * HDIM;
    cvt_f32_bf16<<<(unsigned)((n1 + 255) / 256), 256, 0, stream>>>(hs, hsb, n1);
    size_t n2 = (size_t)DINNER * HDIM;
    cvt_f32_bf16<<<(unsigned)((n2 + 255) / 256), 256, 0, stream>>>(Wqkv, wqkvb, n2);
    size_t n3 = (size_t)HDIM * DGATE;
    cvt_f32_bf16<<<(unsigned)((n3 + 255) / 256), 256, 0, stream>>>(Wout, woutb, n3);
  }

  // 2) B/C coefficients (fp32)
  coeff_kernel<<<MROWS, 64, 0, stream>>>(hs, Wb, Wa, Bc, Cc);

  // 3) GEMM1: xz[8192 x 8192] = hs[8192 x 2560] x W_qkv^T  (WMMA bf16)
  gemm_bf16_wmma<<<dim3(DINNER / BN, MROWS / BM), 256, 0, stream>>>(
      hsb, wqkvb, xz, MROWS, DINNER, HDIM);

  // 4) conv + silu + scan -> Y (fp32)
  scan_kernel<<<dim3(DGATE / 256, BSZ), 256, 0, stream>>>(
      xz, convw, Bc, Cc, Alog, dtb, Yb);

  // 5) RMSNorm + z-gate -> yg (bf16)
  norm_gate_kernel<<<(unsigned)((size_t)MROWS * GN / 2), 256, 0, stream>>>(
      Yb, xz, convw, normw, yg);

  // 6) GEMM2: out[8192 x 2560] = yg[8192 x 4096] x W_out^T  (WMMA bf16)
  gemm_bf16_wmma<<<dim3(HDIM / BN, MROWS / BM), 256, 0, stream>>>(
      yg, woutb, out, MROWS, HDIM, DGATE);
}
